// GCMNetActor_43198781063756
// MI455X (gfx1250) — matrix operand order repaired and checked by reading, not compiled, not run on hardware
//
#include <hip/hip_runtime.h>
#include <hip/hip_bf16.h>

#define DEVI __device__ __forceinline__

typedef __attribute__((ext_vector_type(16))) __bf16 v16bf;
typedef __attribute__((ext_vector_type(8)))  float  v8f;

namespace {

constexpr int B_ = 2048, A_ = 8, D_ = 128, G_ = 128, HEADS_ = 4;
constexpr int H_ = 128, MH_ = 4, NL_ = 2;
constexpr int S_ = D_ + 2 * G_;            // 384
constexpr int FCN_ = 256, ACTN_ = 14;
constexpr int FIN_ = S_ + 5 * H_ + 5 * H_; // 1664

constexpr size_t NBA     = (size_t)A_ * B_;        // 16384
constexpr size_t SZ_XL   = NBA * (size_t)(HEADS_ * G_); // 8388608
constexpr size_t SZ_QKV  = NBA * 4u * (size_t)(3 * H_); // 25165824
constexpr size_t SZ_H    = NBA * (size_t)H_;            // 2097152
constexpr size_t SZ_CAT  = NBA * (size_t)S_;            // 6291456
constexpr size_t SZ_MSK  = (size_t)NL_ * NBA * H_;      // 4194304 per tensor
constexpr size_t SZ_GATE = NBA * 4u * H_;               // 8388608
constexpr size_t SZ_XS   = NBA * 4u * H_;               // 8388608
constexpr size_t SZ_FEAT = NBA * (size_t)FIN_;          // 27262976
constexpr size_t SZ_ATT  = NBA * 4u * H_;               // 8388608

// Arena layout (floats), with stage-lifetime aliasing.
constexpr size_t OFF_R1   = 0;                       // max(2*SZ_XL, SZ_QKV)
constexpr size_t OFF_XL   = OFF_R1;
constexpr size_t OFF_XR   = OFF_R1 + SZ_XL;
constexpr size_t OFF_QKV  = OFF_R1;                  // alias (GAT buffers dead)
constexpr size_t SZ_R1    = SZ_QKV;                  // 25165824 >= 2*SZ_XL? no: 2*SZ_XL=16777216 < 25165824 ok
constexpr size_t OFF_H1   = OFF_R1 + SZ_R1;
constexpr size_t OFF_H2   = OFF_H1 + SZ_H;
constexpr size_t OFF_CAT  = OFF_H2 + SZ_H;
constexpr size_t OFF_R2   = OFF_CAT + SZ_CAT;        // max(4*SZ_MSK+SZ_GATE+SZ_XS, SZ_FEAT)
constexpr size_t OFF_MSH_S= OFF_R2;
constexpr size_t OFF_MSC_S= OFF_R2 + 1 * SZ_MSK;
constexpr size_t OFF_MSH_T= OFF_R2 + 2 * SZ_MSK;
constexpr size_t OFF_MSC_T= OFF_R2 + 3 * SZ_MSK;
constexpr size_t OFF_GATE = OFF_R2 + 4 * SZ_MSK;
constexpr size_t OFF_XS   = OFF_GATE + SZ_GATE;
constexpr size_t OFF_FEAT = OFF_R2;                  // alias (msk/gates/xs dead)
constexpr size_t SZ_R2    = 4 * SZ_MSK + SZ_GATE + SZ_XS; // 33554432 >= SZ_FEAT
constexpr size_t OFF_SH0  = OFF_R2 + SZ_R2;
constexpr size_t OFF_SC0  = OFF_SH0 + SZ_H;
constexpr size_t OFF_SH1  = OFF_SC0 + SZ_H;
constexpr size_t OFF_SC1  = OFF_SH1 + SZ_H;
constexpr size_t OFF_TH0  = OFF_SC1 + SZ_H;
constexpr size_t OFF_TC0  = OFF_TH0 + SZ_H;
constexpr size_t OFF_TH1  = OFF_TC0 + SZ_H;
constexpr size_t OFF_TC1  = OFF_TH1 + SZ_H;
constexpr size_t OFF_OBUF = OFF_TC1 + SZ_H;
constexpr size_t OFF_SATT = OFF_OBUF + SZ_XS;
constexpr size_t OFF_TATT = OFF_SATT + SZ_ATT;
constexpr size_t OFF_X1   = OFF_TATT + SZ_ATT;
constexpr size_t OFF_X2B  = OFF_X1 + NBA * (size_t)FCN_;
constexpr size_t OFF_LOG  = OFF_X2B + NBA * (size_t)FCN_;

DEVI float sigmoidf(float x) { return 1.0f / (1.0f + __expf(-x)); }

// --- WMMA fragment loaders (contiguous-K layouts, fp32 -> bf16 in-register) ---
// A (16x32, MxK): lane m=lane&15 holds 16 contiguous K at kb=16*(lane>=16)
DEVI v16bf load_a_frag(const float* __restrict__ row, int kb) {
  v16bf f;
#pragma unroll
  for (int t = 0; t < 8; ++t) f[t] = (__bf16)row[kb + t];
#pragma unroll
  for (int t = 0; t < 8; ++t) f[t + 8] = (__bf16)row[kb + 8 + t];
  return f;
}
// B (32x16, KxN): lane n=lane&15 holds 16 contiguous K at kb=16*(lane>=16).
// W stored row-major (N,K): B[k][n] = W[n][k] -> lane n reads contiguous W row n.
DEVI v16bf load_b_frag(const float* __restrict__ row, int kb) {
  v16bf f;
#pragma unroll
  for (int t = 0; t < 16; ++t) f[t] = (__bf16)row[kb + t];
  return f;
}
DEVI v16bf zero_frag() {
  v16bf f;
#pragma unroll
  for (int t = 0; t < 16; ++t) f[t] = (__bf16)0.0f;
  return f;
}

// Y = act(X*W^T + bias [+ X2*W2^T + bias2]); fp32 I/O, bf16 WMMA compute.
// Wave tile 32x64 (2x4 of 16x16), block 256 thr = 8 waves -> 128x128 block tile.
template <int RELU>
__global__ __launch_bounds__(256) void wmma_gemm_kernel(
    const float* __restrict__ X, long xRow, long xBatch,
    const float* __restrict__ W, long wBatch,
    const float* __restrict__ bias, long bBatch,
    const float* __restrict__ X2, long x2Row, long x2Batch,
    const float* __restrict__ W2, long w2Batch,
    const float* __restrict__ bias2, long b2Batch,
    float* __restrict__ Y, long yRow, long yBatch,
    int M, int N, int K, int K2)
{
  const int lane = threadIdx.x & 31;
  const int wave = threadIdx.x >> 5;
  const int m0 = blockIdx.y * 128 + (wave >> 1) * 32;
  const int n0 = blockIdx.x * 128 + (wave & 1) * 64;
  if (m0 >= M || n0 >= N) return; // wave-uniform: EXEC stays all-ones for WMMA
  const int z = blockIdx.z;

  const int aRow0 = m0 + (lane & 15);
  const int aRow1 = aRow0 + 16;
  const int kbA = (lane >> 4) << 4;   // 0 or 16
  const int kbB = (lane >> 4) << 4;   // 0 or 16

  v8f acc[2][4] = {};

  // ---- operand 1 ----
  {
    const float* Xb = X + (size_t)z * xBatch;
    const float* Wb = W + (size_t)z * wBatch;
    const float* ar0 = Xb + (size_t)aRow0 * xRow;
    const float* ar1 = Xb + (size_t)aRow1 * xRow;
    const float* brow[4];
    bool bval[4];
#pragma unroll
    for (int j = 0; j < 4; ++j) {
      int n = n0 + j * 16 + (lane & 15);
      bval[j] = (n < N);
      brow[j] = Wb + (size_t)(bval[j] ? n : 0) * K;
    }
    for (int k0 = 0; k0 < K; k0 += 32) {
      if (k0 + 32 < K) {
        __builtin_prefetch(ar0 + k0 + 32, 0, 1);
        __builtin_prefetch(brow[0] + k0 + 32, 0, 1);
      }
      v16bf a0 = load_a_frag(ar0, k0 + kbA);
      v16bf a1 = load_a_frag(ar1, k0 + kbA);
      v16bf b[4];
#pragma unroll
      for (int j = 0; j < 4; ++j)
        b[j] = bval[j] ? load_b_frag(brow[j], k0 + kbB) : zero_frag();
#pragma unroll
      for (int j = 0; j < 4; ++j) {
        acc[0][j] = __builtin_amdgcn_wmma_f32_16x16x32_bf16(
            false, a0, false, b[j], (short)0, acc[0][j], false, false);
        acc[1][j] = __builtin_amdgcn_wmma_f32_16x16x32_bf16(
            false, a1, false, b[j], (short)0, acc[1][j], false, false);
      }
    }
  }
  // ---- optional operand 2 (fused accumulate, e.g. LSTM h*Whh^T) ----
  if (X2 != nullptr) {
    const float* Xb = X2 + (size_t)z * x2Batch;
    const float* Wb = W2 + (size_t)z * w2Batch;
    const float* ar0 = Xb + (size_t)aRow0 * x2Row;
    const float* ar1 = Xb + (size_t)aRow1 * x2Row;
    const float* brow[4];
    bool bval[4];
#pragma unroll
    for (int j = 0; j < 4; ++j) {
      int n = n0 + j * 16 + (lane & 15);
      bval[j] = (n < N);
      brow[j] = Wb + (size_t)(bval[j] ? n : 0) * K2;
    }
    for (int k0 = 0; k0 < K2; k0 += 32) {
      v16bf a0 = load_a_frag(ar0, k0 + kbA);
      v16bf a1 = load_a_frag(ar1, k0 + kbA);
      v16bf b[4];
#pragma unroll
      for (int j = 0; j < 4; ++j)
        b[j] = bval[j] ? load_b_frag(brow[j], k0 + kbB) : zero_frag();
#pragma unroll
      for (int j = 0; j < 4; ++j) {
        acc[0][j] = __builtin_amdgcn_wmma_f32_16x16x32_bf16(
            false, a0, false, b[j], (short)0, acc[0][j], false, false);
        acc[1][j] = __builtin_amdgcn_wmma_f32_16x16x32_bf16(
            false, a1, false, b[j], (short)0, acc[1][j], false, false);
      }
    }
  }
  // ---- epilogue: C layout = VGPR r: M=m0+i*16+r+8*(lane>=16), N=n0+j*16+(lane&15)
  float* Yb = Y + (size_t)z * yBatch;
  const float* bp  = bias  ? bias  + (size_t)z * bBatch  : nullptr;
  const float* bp2 = bias2 ? bias2 + (size_t)z * b2Batch : nullptr;
#pragma unroll
  for (int i = 0; i < 2; ++i) {
    const size_t rowBase = (size_t)(m0 + i * 16 + ((lane >> 4) << 3));
#pragma unroll
    for (int j = 0; j < 4; ++j) {
      const int col = n0 + j * 16 + (lane & 15);
      if (col >= N) continue;
      float ba = (bp ? bp[col] : 0.0f) + (bp2 ? bp2[col] : 0.0f);
      float* yp = Yb + rowBase * yRow + col;
#pragma unroll
      for (int r = 0; r < 8; ++r) {
        float v = acc[i][j][r] + ba;
        if (RELU) v = fmaxf(v, 0.0f);
        yp[(size_t)r * yRow] = v;
      }
    }
  }
}

// --- GATv2 pairwise attention: one block per batch element b ---
__global__ __launch_bounds__(256) void gat_attn_kernel(
    const float* __restrict__ xl, const float* __restrict__ xr,
    const float* __restrict__ att, const float* __restrict__ bias,
    float* __restrict__ out)
{
  __shared__ float sxl[A_ * HEADS_ * G_]; // 4096
  __shared__ float sxr[A_ * HEADS_ * G_];
  __shared__ float swa[HEADS_ * G_];      // 512
  __shared__ float se[A_ * A_ * HEADS_];  // 256
  const int b = blockIdx.x, tid = threadIdx.x;
  for (int p = tid; p < A_ * HEADS_ * G_; p += 256) {
    sxl[p] = xl[(size_t)b * A_ * (HEADS_ * G_) + p];
    sxr[p] = xr[(size_t)b * A_ * (HEADS_ * G_) + p];
  }
  for (int p = tid; p < HEADS_ * G_; p += 256) swa[p] = att[p];
  __syncthreads();
  { // e[b,i,j,h] = sum_c leaky(xl[j,h,c] + xr[i,h,c]) * att[h,c]
    const int h = tid & 3, j = (tid >> 2) & 7, i = tid >> 5;
    const float* pl = sxl + (j * HEADS_ + h) * G_;
    const float* pr = sxr + (i * HEADS_ + h) * G_;
    const float* pa = swa + h * G_;
    float acc = 0.0f;
    for (int c = 0; c < G_; ++c) {
      float s = pl[c] + pr[c];
      s = s > 0.0f ? s : 0.2f * s;
      acc += s * pa[c];
    }
    se[(i * A_ + j) * HEADS_ + h] = acc;
  }
  __syncthreads();
  if (tid < A_ * HEADS_) { // softmax over j
    const int h = tid & 3, i = tid >> 2;
    float mx = -1e30f;
    for (int j = 0; j < A_; ++j) mx = fmaxf(mx, se[(i * A_ + j) * HEADS_ + h]);
    float ex[A_], ssum = 0.0f;
    for (int j = 0; j < A_; ++j) { ex[j] = __expf(se[(i * A_ + j) * HEADS_ + h] - mx); ssum += ex[j]; }
    float inv = 1.0f / ssum;
    for (int j = 0; j < A_; ++j) se[(i * A_ + j) * HEADS_ + h] = ex[j] * inv;
  }
  __syncthreads();
  for (int p = tid; p < A_ * G_; p += 256) { // out[i,c] = mean_h sum_j a*xl + bias, relu
    const int c = p & (G_ - 1), i = p >> 7;
    float acc = 0.0f;
    for (int h = 0; h < HEADS_; ++h) {
      float s = 0.0f;
      for (int j = 0; j < A_; ++j)
        s += se[(i * A_ + j) * HEADS_ + h] * sxl[(j * HEADS_ + h) * G_ + c];
      acc += s;
    }
    acc = acc * 0.25f + bias[c];
    out[((size_t)b * A_ + i) * G_ + c] = fmaxf(acc, 0.0f);
  }
}

__global__ void concat_cat_kernel(const float* __restrict__ obs,
                                  const float* __restrict__ h1,
                                  const float* __restrict__ h2,
                                  float* __restrict__ cat, int total)
{
  int i = blockIdx.x * blockDim.x + threadIdx.x;
  if (i >= total) return;
  int c = i % S_; size_t r = (size_t)(i / S_);
  float v;
  if (c < D_)           v = obs[r * D_ + c];
  else if (c < D_ + G_) v = h1[r * G_ + (c - D_)];
  else                  v = h2[r * G_ + (c - D_ - G_)];
  cat[i] = v;
}

// src (B,A,NL,H) * mask(B,A) -> dst [l][a][b][h]
__global__ void mask_states_kernel(const float* __restrict__ src,
                                   const float* __restrict__ mask,
                                   float* __restrict__ dst, int total)
{
  int i = blockIdx.x * blockDim.x + threadIdx.x;
  if (i >= total) return;
  int h = i % H_; int t = i / H_;
  int l = t % NL_; t /= NL_;
  int a = t % A_; int b = t / A_;
  float m = mask[b * A_ + a];
  dst[(((size_t)l * A_ + a) * B_ + b) * H_ + h] = src[i] * m;
}

// gates [a][b][4H] (i,f,g,o), cprev [a][b][H] -> hout/cout [a][b][H]
__global__ void lstm_pointwise_kernel(const float* __restrict__ gates,
                                      const float* __restrict__ cprev,
                                      float* __restrict__ hout,
                                      float* __restrict__ cout, int total)
{
  int i = blockIdx.x * blockDim.x + threadIdx.x;
  if (i >= total) return;
  int h = i % H_; size_t row = (size_t)(i / H_);
  const float* g = gates + row * (4 * H_);
  float ig = sigmoidf(g[h]);
  float fg = sigmoidf(g[H_ + h]);
  float gg = tanhf(g[2 * H_ + h]);
  float og = sigmoidf(g[3 * H_ + h]);
  float c2 = fg * cprev[i] + ig * gg;
  cout[i] = c2;
  hout[i] = og * tanhf(c2);
}

// xs[a][b][t][H], t order = (h0, h1, c0, c1)
__global__ void stack4_kernel(const float* __restrict__ h0,
                              const float* __restrict__ h1,
                              const float* __restrict__ c0,
                              const float* __restrict__ c1,
                              float* __restrict__ xs, int total)
{
  int i = blockIdx.x * blockDim.x + threadIdx.x;
  if (i >= total) return;
  int h = i % H_; size_t ab = (size_t)(i / H_);
  float* p = xs + ab * 4 * H_ + h;
  p[0]      = h0[i];
  p[H_]     = h1[i];
  p[2 * H_] = c0[i];
  p[3 * H_] = c1[i];
}

// 4-token MHA core: one thread per (a,b,head); qkv rows [ab*4+t][384]
__global__ void mha_attn_kernel(const float* __restrict__ qkv, float* __restrict__ obuf)
{
  int i = blockIdx.x * blockDim.x + threadIdx.x;
  if (i >= A_ * B_ * MH_) return;
  const int hd = i % MH_;
  const size_t ab = (size_t)(i / MH_);
  const float* base = qkv + ab * 4 * (3 * H_);
  const int qo = hd * 32, ko = H_ + hd * 32, vo = 2 * H_ + hd * 32;
  const float scale = 0.17677669529663687f; // 1/sqrt(32)
  float s[4][4];
#pragma unroll
  for (int t = 0; t < 4; ++t)
#pragma unroll
    for (int u = 0; u < 4; ++u) {
      float a = 0.0f;
      for (int d = 0; d < 32; ++d)
        a += base[t * (3 * H_) + qo + d] * base[u * (3 * H_) + ko + d];
      s[t][u] = a * scale;
    }
#pragma unroll
  for (int t = 0; t < 4; ++t) {
    float mx = fmaxf(fmaxf(s[t][0], s[t][1]), fmaxf(s[t][2], s[t][3]));
    float e0 = __expf(s[t][0] - mx), e1 = __expf(s[t][1] - mx);
    float e2 = __expf(s[t][2] - mx), e3 = __expf(s[t][3] - mx);
    float inv = 1.0f / (e0 + e1 + e2 + e3);
    s[t][0] = e0 * inv; s[t][1] = e1 * inv; s[t][2] = e2 * inv; s[t][3] = e3 * inv;
  }
#pragma unroll
  for (int t = 0; t < 4; ++t)
    for (int d = 0; d < 32; ++d) {
      float a = s[t][0] * base[0 * (3 * H_) + vo + d]
              + s[t][1] * base[1 * (3 * H_) + vo + d]
              + s[t][2] * base[2 * (3 * H_) + vo + d]
              + s[t][3] * base[3 * (3 * H_) + vo + d];
      obuf[(ab * 4 + t) * H_ + hd * 32 + d] = a;
    }
}

// feat[a][b][1664] = [cat(b,a,384), sh1(128), satt(512), th1(128), tatt(512)]
__global__ void feat_kernel(const float* __restrict__ cat,
                            const float* __restrict__ sh1,
                            const float* __restrict__ satt,
                            const float* __restrict__ th1,
                            const float* __restrict__ tatt,
                            float* __restrict__ feat, int total)
{
  int i = blockIdx.x * blockDim.x + threadIdx.x;
  if (i >= total) return;
  int c = i % FIN_; size_t ab = (size_t)(i / FIN_);
  int b = (int)(ab % B_), a = (int)(ab / B_);
  float v;
  if (c < S_)            v = cat[((size_t)b * A_ + a) * S_ + c];
  else if (c < S_ + H_)  v = sh1[ab * H_ + (c - S_)];
  else if (c < S_ + 5*H_) v = satt[ab * (4 * H_) + (c - S_ - H_)];
  else if (c < S_ + 6*H_) v = th1[ab * H_ + (c - S_ - 5 * H_)];
  else                    v = tatt[ab * (4 * H_) + (c - S_ - 6 * H_)];
  feat[i] = v;
}

// row = a*B+b of logits[16384][14] -> d_out[(b*A+a)*14 + j] = log_softmax
__global__ void logsoftmax_kernel(const float* __restrict__ logits, float* __restrict__ out)
{
  int row = blockIdx.x * blockDim.x + threadIdx.x;
  if (row >= (int)NBA) return;
  const float* p = logits + (size_t)row * ACTN_;
  float mx = -1e30f;
  for (int j = 0; j < ACTN_; ++j) mx = fmaxf(mx, p[j]);
  float s = 0.0f;
  for (int j = 0; j < ACTN_; ++j) s += __expf(p[j] - mx);
  float lse = mx + __logf(s);
  int a = row / B_, b = row % B_;
  float* q = out + ((size_t)b * A_ + a) * ACTN_;
  for (int j = 0; j < ACTN_; ++j) q[j] = p[j] - lse;
}

static inline int cdiv(int a, int b) { return (a + b - 1) / b; }

static void gemm(hipStream_t st,
                 const float* X, long xRow, long xBatch,
                 const float* W, long wBatch,
                 const float* bias, long bBatch,
                 const float* X2, long x2Row, long x2Batch,
                 const float* W2, long w2Batch,
                 const float* bias2, long b2Batch,
                 float* Y, long yRow, long yBatch,
                 int M, int N, int K, int K2, int nbatch, bool relu)
{
  dim3 grid(cdiv(N, 128), cdiv(M, 128), nbatch);
  dim3 blk(256);
  if (relu)
    wmma_gemm_kernel<1><<<grid, blk, 0, st>>>(X, xRow, xBatch, W, wBatch, bias, bBatch,
                                              X2, x2Row, x2Batch, W2, w2Batch, bias2, b2Batch,
                                              Y, yRow, yBatch, M, N, K, K2);
  else
    wmma_gemm_kernel<0><<<grid, blk, 0, st>>>(X, xRow, xBatch, W, wBatch, bias, bBatch,
                                              X2, x2Row, x2Batch, W2, w2Batch, bias2, b2Batch,
                                              Y, yRow, yBatch, M, N, K, K2);
}

} // namespace

extern "C" void kernel_launch(void* const* d_in, const int* in_sizes, int n_in,
                              void* d_out, int out_size, void* d_ws, size_t ws_size,
                              hipStream_t stream)
{
  (void)in_sizes; (void)n_in; (void)out_size; (void)ws_size;
  const float* obs    = (const float*)d_in[0];
  const float* somu_h = (const float*)d_in[1];
  const float* somu_c = (const float*)d_in[2];
  const float* scmu_h = (const float*)d_in[3];
  const float* scmu_c = (const float*)d_in[4];
  const float* masks  = (const float*)d_in[5];
  const float* gWl0 = (const float*)d_in[6];   const float* gbl0 = (const float*)d_in[7];
  const float* gWr0 = (const float*)d_in[8];   const float* gbr0 = (const float*)d_in[9];
  const float* gat0 = (const float*)d_in[10];  const float* gbs0 = (const float*)d_in[11];
  const float* gWl1 = (const float*)d_in[12];  const float* gbl1 = (const float*)d_in[13];
  const float* gWr1 = (const float*)d_in[14];  const float* gbr1 = (const float*)d_in[15];
  const float* gat1 = (const float*)d_in[16];  const float* gbs1 = (const float*)d_in[17];
  const float* sWih0 = (const float*)d_in[18]; const float* sWhh0 = (const float*)d_in[19];
  const float* sbih0 = (const float*)d_in[20]; const float* sbhh0 = (const float*)d_in[21];
  const float* sWih1 = (const float*)d_in[22]; const float* sWhh1 = (const float*)d_in[23];
  const float* sbih1 = (const float*)d_in[24]; const float* sbhh1 = (const float*)d_in[25];
  const float* sqkvw = (const float*)d_in[26]; const float* sqkvb = (const float*)d_in[27];
  const float* soutw = (const float*)d_in[28]; const float* soutb = (const float*)d_in[29];
  const float* tWih0 = (const float*)d_in[30]; const float* tWhh0 = (const float*)d_in[31];
  const float* tbih0 = (const float*)d_in[32]; const float* tbhh0 = (const float*)d_in[33];
  const float* tWih1 = (const float*)d_in[34]; const float* tWhh1 = (const float*)d_in[35];
  const float* tbih1 = (const float*)d_in[36]; const float* tbhh1 = (const float*)d_in[37];
  const float* tqkvw = (const float*)d_in[38]; const float* tqkvb = (const float*)d_in[39];
  const float* toutw = (const float*)d_in[40]; const float* toutb = (const float*)d_in[41];
  const float* fcw0 = (const float*)d_in[42];  const float* fcb0 = (const float*)d_in[43];
  const float* fcw1 = (const float*)d_in[44];  const float* fcb1 = (const float*)d_in[45];
  const float* actw = (const float*)d_in[46];  const float* actb = (const float*)d_in[47];

  float* ws = (float*)d_ws;
  const int M_ALL = (int)NBA;   // 16384
  const int totH  = (int)(NBA * H_);       // 2097152
  const int totMS = B_ * A_ * NL_ * H_;    // 4194304
  const int totCat = (int)(NBA * S_);
  const int totFeat = (int)SZ_FEAT;

  // ======== GAT layer 0 ========
  gemm(stream, obs, D_, 0, gWl0, 0, gbl0, 0,
       nullptr, 0, 0, nullptr, 0, nullptr, 0,
       ws + OFF_XL, HEADS_ * G_, 0, M_ALL, HEADS_ * G_, D_, 0, 1, false);
  gemm(stream, obs, D_, 0, gWr0, 0, gbr0, 0,
       nullptr, 0, 0, nullptr, 0, nullptr, 0,
       ws + OFF_XR, HEADS_ * G_, 0, M_ALL, HEADS_ * G_, D_, 0, 1, false);
  gat_attn_kernel<<<B_, 256, 0, stream>>>(ws + OFF_XL, ws + OFF_XR, gat0, gbs0, ws + OFF_H1);

  // ======== GAT layer 1 ========
  gemm(stream, ws + OFF_H1, G_, 0, gWl1, 0, gbl1, 0,
       nullptr, 0, 0, nullptr, 0, nullptr, 0,
       ws + OFF_XL, HEADS_ * G_, 0, M_ALL, HEADS_ * G_, G_, 0, 1, false);
  gemm(stream, ws + OFF_H1, G_, 0, gWr1, 0, gbr1, 0,
       nullptr, 0, 0, nullptr, 0, nullptr, 0,
       ws + OFF_XR, HEADS_ * G_, 0, M_ALL, HEADS_ * G_, G_, 0, 1, false);
  gat_attn_kernel<<<B_, 256, 0, stream>>>(ws + OFF_XL, ws + OFF_XR, gat1, gbs1, ws + OFF_H2);

  // ======== gnn_cat = [obs, h1, h2] ========
  concat_cat_kernel<<<cdiv(totCat, 256), 256, 0, stream>>>(obs, ws + OFF_H1, ws + OFF_H2,
                                                           ws + OFF_CAT, totCat);

  // ======== masked recurrent states -> [l][a][b][h] ========
  mask_states_kernel<<<cdiv(totMS, 256), 256, 0, stream>>>(somu_h, masks, ws + OFF_MSH_S, totMS);
  mask_states_kernel<<<cdiv(totMS, 256), 256, 0, stream>>>(somu_c, masks, ws + OFF_MSC_S, totMS);
  mask_states_kernel<<<cdiv(totMS, 256), 256, 0, stream>>>(scmu_h, masks, ws + OFF_MSH_T, totMS);
  mask_states_kernel<<<cdiv(totMS, 256), 256, 0, stream>>>(scmu_c, masks, ws + OFF_MSC_T, totMS);

  const long lOff = (long)(NBA * H_); // layer stride inside masked-state tensor

  // ======== somu LSTM L0 (fused x*Wih^T + h*Whh^T) ========
  gemm(stream, obs, A_ * D_, D_, sWih0, 4 * H_ * D_, sbih0, 4 * H_,
       ws + OFF_MSH_S, H_, (long)(B_ * H_), sWhh0, 4 * H_ * H_, sbhh0, 4 * H_,
       ws + OFF_GATE, 4 * H_, (long)(B_ * 4 * H_), B_, 4 * H_, D_, H_, A_, false);
  lstm_pointwise_kernel<<<cdiv(totH, 256), 256, 0, stream>>>(
      ws + OFF_GATE, ws + OFF_MSC_S, ws + OFF_SH0, ws + OFF_SC0, totH);
  // ======== somu LSTM L1 ========
  gemm(stream, ws + OFF_SH0, H_, (long)(B_ * H_), sWih1, 4 * H_ * H_, sbih1, 4 * H_,
       ws + OFF_MSH_S + lOff, H_, (long)(B_ * H_), sWhh1, 4 * H_ * H_, sbhh1, 4 * H_,
       ws + OFF_GATE, 4 * H_, (long)(B_ * 4 * H_), B_, 4 * H_, H_, H_, A_, false);
  lstm_pointwise_kernel<<<cdiv(totH, 256), 256, 0, stream>>>(
      ws + OFF_GATE, ws + OFF_MSC_S + lOff, ws + OFF_SH1, ws + OFF_SC1, totH);
  // ======== somu MHA ========
  stack4_kernel<<<cdiv(totH, 256), 256, 0, stream>>>(
      ws + OFF_SH0, ws + OFF_SH1, ws + OFF_SC0, ws + OFF_SC1, ws + OFF_XS, totH);
  gemm(stream, ws + OFF_XS, H_, (long)(B_ * 4 * H_), sqkvw, 3 * H_ * H_, sqkvb, 3 * H_,
       nullptr, 0, 0, nullptr, 0, nullptr, 0,
       ws + OFF_QKV, 3 * H_, (long)(B_ * 4 * 3 * H_), B_ * 4, 3 * H_, H_, 0, A_, false);
  mha_attn_kernel<<<cdiv(A_ * B_ * MH_, 256), 256, 0, stream>>>(ws + OFF_QKV, ws + OFF_OBUF);
  gemm(stream, ws + OFF_OBUF, H_, (long)(B_ * 4 * H_), soutw, H_ * H_, soutb, H_,
       nullptr, 0, 0, nullptr, 0, nullptr, 0,
       ws + OFF_SATT, H_, (long)(B_ * 4 * H_), B_ * 4, H_, H_, 0, A_, false);

  // ======== scmu LSTM L0 (input = gnn_cat, K=384) ========
  gemm(stream, ws + OFF_CAT, A_ * S_, S_, tWih0, 4 * H_ * S_, tbih0, 4 * H_,
       ws + OFF_MSH_T, H_, (long)(B_ * H_), tWhh0, 4 * H_ * H_, tbhh0, 4 * H_,
       ws + OFF_GATE, 4 * H_, (long)(B_ * 4 * H_), B_, 4 * H_, S_, H_, A_, false);
  lstm_pointwise_kernel<<<cdiv(totH, 256), 256, 0, stream>>>(
      ws + OFF_GATE, ws + OFF_MSC_T, ws + OFF_TH0, ws + OFF_TC0, totH);
  // ======== scmu LSTM L1 ========
  gemm(stream, ws + OFF_TH0, H_, (long)(B_ * H_), tWih1, 4 * H_ * H_, tbih1, 4 * H_,
       ws + OFF_MSH_T + lOff, H_, (long)(B_ * H_), tWhh1, 4 * H_ * H_, tbhh1, 4 * H_,
       ws + OFF_GATE, 4 * H_, (long)(B_ * 4 * H_), B_, 4 * H_, H_, H_, A_, false);
  lstm_pointwise_kernel<<<cdiv(totH, 256), 256, 0, stream>>>(
      ws + OFF_GATE, ws + OFF_MSC_T + lOff, ws + OFF_TH1, ws + OFF_TC1, totH);
  // ======== scmu MHA ========
  stack4_kernel<<<cdiv(totH, 256), 256, 0, stream>>>(
      ws + OFF_TH0, ws + OFF_TH1, ws + OFF_TC0, ws + OFF_TC1, ws + OFF_XS, totH);
  gemm(stream, ws + OFF_XS, H_, (long)(B_ * 4 * H_), tqkvw, 3 * H_ * H_, tqkvb, 3 * H_,
       nullptr, 0, 0, nullptr, 0, nullptr, 0,
       ws + OFF_QKV, 3 * H_, (long)(B_ * 4 * 3 * H_), B_ * 4, 3 * H_, H_, 0, A_, false);
  mha_attn_kernel<<<cdiv(A_ * B_ * MH_, 256), 256, 0, stream>>>(ws + OFF_QKV, ws + OFF_OBUF);
  gemm(stream, ws + OFF_OBUF, H_, (long)(B_ * 4 * H_), toutw, H_ * H_, toutb, H_,
       nullptr, 0, 0, nullptr, 0, nullptr, 0,
       ws + OFF_TATT, H_, (long)(B_ * 4 * H_), B_ * 4, H_, H_, 0, A_, false);

  // ======== feature gather + MLP head ========
  feat_kernel<<<cdiv(totFeat, 256), 256, 0, stream>>>(
      ws + OFF_CAT, ws + OFF_SH1, ws + OFF_SATT, ws + OFF_TH1, ws + OFF_TATT,
      ws + OFF_FEAT, totFeat);
  gemm(stream, ws + OFF_FEAT, FIN_, 0, fcw0, 0, fcb0, 0,
       nullptr, 0, 0, nullptr, 0, nullptr, 0,
       ws + OFF_X1, FCN_, 0, M_ALL, FCN_, FIN_, 0, 1, true);
  gemm(stream, ws + OFF_X1, FCN_, 0, fcw1, 0, fcb1, 0,
       nullptr, 0, 0, nullptr, 0, nullptr, 0,
       ws + OFF_X2B, FCN_, 0, M_ALL, FCN_, FCN_, 0, 1, true);
  gemm(stream, ws + OFF_X2B, FCN_, 0, actw, 0, actb, 0,
       nullptr, 0, 0, nullptr, 0, nullptr, 0,
       ws + OFF_LOG, ACTN_, 0, M_ALL, ACTN_, FCN_, 0, 1, false);
  logsoftmax_kernel<<<cdiv(M_ALL, 256), 256, 0, stream>>>(ws + OFF_LOG, (float*)d_out);
}
